// MultiLatentAttention_40097814675887
// MI455X (gfx1250) — matrix operand order repaired and checked by reading, not compiled, not run on hardware
//
#include <hip/hip_runtime.h>

// ---------------------------------------------------------------------------
// MLA forward for MI455X (gfx1250, wave32, WMMA).
// GEMMs: bf16 WMMA, W tiles double-buffered in LDS via async global->LDS DMA.
// Attention: flash-style; 4 waves/block share one head's K/V stream, staged
// in double-buffered LDS by async DMA (ASYNCcnt), WMMA QK^T / PV from LDS.
// ---------------------------------------------------------------------------

typedef __bf16 bf16_t;
typedef __bf16 v8bf  __attribute__((ext_vector_type(8)));
typedef __bf16 v16bf __attribute__((ext_vector_type(16)));
typedef float  v8f   __attribute__((ext_vector_type(8)));

#define WMMA_BF16(a, b, c) \
  __builtin_amdgcn_wmma_f32_16x16x32_bf16(false, (a), false, (b), (short)0, (c), false, false)

// Problem constants (match reference)
#define S_LEN   2048
#define H_DIM   4096
#define NHEAD   32
#define Q_LORA  1536
#define KV_LORA 512
#define Q_HEAD  192   // 128 nope + 64 rope
#define V_HEAD  128
#define KV_UP_N 8192  // NH * (128 + 128)
#define Q_UP_N  6144  // NH * 192
#define KVC_N   576   // KV_LORA + QK_ROPE

static __device__ __forceinline__ unsigned short f32_to_bf16_rne(float f) {
  unsigned int u = __builtin_bit_cast(unsigned int, f);
  u += 0x7FFFu + ((u >> 16) & 1u);
  return (unsigned short)(u >> 16);
}
static __device__ __forceinline__ bf16_t to_bf(float f) {
  unsigned short s = f32_to_bf16_rne(f);
  return __builtin_bit_cast(bf16_t, s);
}
static __device__ __forceinline__ v8bf ldg8(const bf16_t* p) {
  return *reinterpret_cast<const v8bf*>(p);
}
static __device__ __forceinline__ v8bf lds8(const unsigned short* p) {
  return *reinterpret_cast<const v8bf*>(p);
}
static __device__ __forceinline__ v16bf cat8(v8bf lo, v8bf hi) {
  return __builtin_shufflevector(lo, hi, 0,1,2,3,4,5,6,7,8,9,10,11,12,13,14,15);
}
static __device__ __forceinline__ v8f vzero8() {
  v8f z = {0.f,0.f,0.f,0.f,0.f,0.f,0.f,0.f};
  return z;
}

// Async global -> LDS DMA (16B per lane). Generic pointers to __shared__ live
// in the LDS aperture: addr[31:0] IS the LDS byte offset (ISA 10.2 aperture
// mapping), so the low dword of the pointer is the VDST LDS address.
static __device__ __forceinline__ void async_copy_b128(const void* gptr, void* lptr) {
  const unsigned loff = (unsigned)(unsigned long long)lptr;
  asm volatile("global_load_async_to_lds_b128 %0, %1, off"
               :: "v"(loff), "v"(gptr) : "memory");
}

// -------------------------- fp32 -> bf16 convert ---------------------------
__global__ void k_f32_to_bf16(const float* __restrict__ in, bf16_t* __restrict__ out, int n) {
  int i = blockIdx.x * blockDim.x + threadIdx.x;
  if (i < n) out[i] = to_bf(in[i]);
}

// ------------------------------ WMMA GEMM ----------------------------------
// C[M,N] = A[M,K] * W[N,K]^T + bias.
// Block = 4 waves -> 128x64 tile; the 64x32 W k-slab is shared by all waves,
// double-buffered in LDS and filled by global_load_async_to_lds_b128 while
// the previous slab is consumed (s_wait_asynccnt + barrier handshake).
#define GEMM_THREADS 128
template <bool OUT_F32>
__global__ __launch_bounds__(GEMM_THREADS)
void k_gemm_wmma(const bf16_t* __restrict__ A, int lda,
                 const bf16_t* __restrict__ W,
                 const float*  __restrict__ bias,
                 void* __restrict__ Cout,
                 int M, int N, int K) {
  __shared__ __align__(16) unsigned short sW[2][64 * 32];   // 2 x 4KB bf16 tiles

  const int ntN  = N >> 6;
  const int n0   = (blockIdx.x % ntN) << 6;
  const int mblk = (blockIdx.x / ntN) << 7;
  const int w    = threadIdx.x >> 5;
  const int lane = threadIdx.x & 31;
  const int m0   = mblk + (w << 5);
  const int h    = lane >> 4;
  const int l16  = lane & 15;

  auto stage = [&](int p, int kk) {
    #pragma unroll
    for (int u = 0; u < 2; ++u) {
      const int ch  = (int)threadIdx.x + u * GEMM_THREADS;  // 256 x 16B chunks
      const int row = ch >> 2;
      const int c8  = (ch & 3) << 3;
      async_copy_b128(W + (size_t)(n0 + row) * K + kk + c8, &sW[p][ch * 8]);
    }
  };

  v8f acc[2][4];
  #pragma unroll
  for (int mt = 0; mt < 2; ++mt)
    #pragma unroll
    for (int c = 0; c < 4; ++c) acc[mt][c] = vzero8();

  const int nk = K >> 5;
  stage(0, 0);
  for (int i = 0; i < nk; ++i) {
    const int kk = i << 5;

    // A-fragment loads are independent of the LDS slab: issue them first so
    // their latency overlaps the DMA handshake below.
    const bf16_t* pa0 = A + (size_t)(m0 + l16) * lda + kk;
    const bf16_t* pa1 = pa0 + (size_t)16 * lda;
    __builtin_prefetch(pa0 + 64, 0, 1);
    v16bf a0 = cat8(ldg8(pa0 + h * 8), ldg8(pa0 + 16 + h * 8));
    v16bf a1 = cat8(ldg8(pa1 + h * 8), ldg8(pa1 + 16 + h * 8));

    if (i + 1 < nk) {
      stage((i + 1) & 1, kk + 32);                 // DMA next slab (in flight)
      asm volatile("s_wait_asynccnt 0x2" ::: "memory");  // own part of slab i landed
    } else {
      asm volatile("s_wait_asynccnt 0x0" ::: "memory");
    }
    __syncthreads();                               // everyone's part landed

    const unsigned short* sw = sW[i & 1];
    #pragma unroll
    for (int c = 0; c < 4; ++c) {
      const unsigned short* pw = sw + (c * 16 + l16) * 32 + h * 8;
      v16bf b = cat8(lds8(pw), lds8(pw + 16));
      acc[0][c] = WMMA_BF16(a0, b, acc[0][c]);
      acc[1][c] = WMMA_BF16(a1, b, acc[1][c]);
    }
    __syncthreads();                               // reads done before slab reuse
  }

  #pragma unroll
  for (int mt = 0; mt < 2; ++mt) {
    #pragma unroll
    for (int c = 0; c < 4; ++c) {
      const int col = n0 + c * 16 + l16;
      const float bv = bias ? bias[col] : 0.f;
      #pragma unroll
      for (int r = 0; r < 8; ++r) {
        const int row = m0 + mt * 16 + h * 8 + r;
        const float v = acc[mt][c][r] + bv;
        if constexpr (OUT_F32)
          reinterpret_cast<float*>(Cout)[(size_t)row * N + col] = v;
        else
          reinterpret_cast<bf16_t*>(Cout)[(size_t)row * N + col] = to_bf(v);
      }
    }
  }
}

// ----------------------------- layout kernels ------------------------------
__global__ void k_q_layout(const bf16_t* __restrict__ qfull, bf16_t* __restrict__ Qbuf) {
  int i = blockIdx.x * blockDim.x + threadIdx.x;
  if (i >= NHEAD * S_LEN * Q_HEAD) return;
  int d = i % Q_HEAD; int t = i / Q_HEAD;
  int s = t % S_LEN;  int hh = t / S_LEN;
  Qbuf[i] = qfull[(size_t)s * Q_UP_N + hh * Q_HEAD + d];
}
__global__ void k_k_layout(const bf16_t* __restrict__ kvu, const bf16_t* __restrict__ kvc,
                           bf16_t* __restrict__ Kbuf) {
  int i = blockIdx.x * blockDim.x + threadIdx.x;
  if (i >= NHEAD * S_LEN * Q_HEAD) return;
  int d = i % Q_HEAD; int t = i / Q_HEAD;
  int s = t % S_LEN;  int hh = t / S_LEN;
  bf16_t v;
  if (d < 128) v = kvu[(size_t)s * KV_UP_N + hh * 256 + d];
  else         v = kvc[(size_t)s * KVC_N + KV_LORA + (d - 128)];
  Kbuf[i] = v;
}
__global__ void k_v_layout(const bf16_t* __restrict__ kvu, bf16_t* __restrict__ Vt) {
  int i = blockIdx.x * blockDim.x + threadIdx.x;
  if (i >= NHEAD * V_HEAD * S_LEN) return;
  int s = i % S_LEN; int t = i / S_LEN;
  int v = t % V_HEAD; int hh = t / V_HEAD;
  Vt[i] = kvu[(size_t)s * KV_UP_N + hh * 256 + 128 + v];
}

// --------------------------- flash attention -------------------------------
// Block = 4 waves on ONE head, 64 consecutive queries (16/wave). All waves
// share the head's K/V stream: each 32-key chunk of K (32x192, contiguous
// 12KB of Kbuf) and V (128x32 slice of Vt) is DMA'd into double-buffered LDS
// with global_load_async_to_lds_b128 while the previous chunk is consumed.
// QK^T and PV B-fragments are then ds_load_b128 reads (short latency, no
// 64-bit address chains in the hot loop; 4x cut in global K/V traffic).
#define ATT_THREADS 128
#define ATT_WAVES   4
__global__ __launch_bounds__(ATT_THREADS)
void k_attention(const bf16_t* __restrict__ Qbuf, const bf16_t* __restrict__ Kbuf,
                 const bf16_t* __restrict__ Vt, float* __restrict__ ohead) {
  __shared__ __align__(16) unsigned short sK[2][32 * Q_HEAD];   // 2 x 12KB
  __shared__ __align__(16) unsigned short sV[2][V_HEAD * 32];   // 2 x 8KB
  __shared__ __align__(16) float          sS[ATT_WAVES][16 * 32];
  __shared__ __align__(16) unsigned short sP[ATT_WAVES][16 * 32];
  __shared__ __align__(16) float          sC[ATT_WAVES][16];

  const int w    = threadIdx.x >> 5;
  const int lane = threadIdx.x & 31;
  const int hh   = blockIdx.x >> 5;                 // 32 q-blocks per head
  const int q0   = ((blockIdx.x & 31) << 6) + (w << 4);
  const int h    = lane >> 4;
  const int l16  = lane & 15;
  const float scale = 0.0721687836487032f;          // 1/sqrt(192)

  const bf16_t* kb_head = Kbuf + (size_t)hh * S_LEN * Q_HEAD;
  const bf16_t* vt_head = Vt + (size_t)hh * V_HEAD * S_LEN;

  // DMA one 32-key chunk: K is a flat contiguous 12KB block; V is 128 rows
  // of 64B from the transposed layout.
  auto stage = [&](int p, int kb) {
    const bf16_t* ksrc = kb_head + (size_t)kb * Q_HEAD;   // contiguous tile
    #pragma unroll
    for (int u = 0; u < 6; ++u) {                          // 768 x 16B chunks
      const int ch = (int)threadIdx.x + u * ATT_THREADS;
      async_copy_b128(ksrc + ch * 8, &sK[p][ch * 8]);
    }
    #pragma unroll
    for (int u = 0; u < 4; ++u) {                          // 512 x 16B chunks
      const int ch  = (int)threadIdx.x + u * ATT_THREADS;
      const int row = ch >> 2;
      const int c8  = (ch & 3) << 3;
      async_copy_b128(vt_head + (size_t)row * S_LEN + kb + c8, &sV[p][ch * 8]);
    }
  };

  // Q A-fragments (resident in registers for the whole key stream)
  v16bf aq[6];
  const bf16_t* qrow = Qbuf + ((size_t)hh * S_LEN + q0 + l16) * Q_HEAD;
  #pragma unroll
  for (int t = 0; t < 6; ++t) {
    const bf16_t* p = qrow + t * 32;
    aq[t] = cat8(ldg8(p + h * 8), ldg8(p + 16 + h * 8));
  }

  v8f oacc[8];
  #pragma unroll
  for (int c = 0; c < 8; ++c) oacc[c] = vzero8();
  float m_run = -1e30f, l_run = 0.f;

  const int nkb = S_LEN / 32;
  stage(0, 0);
  for (int i = 0; i < nkb; ++i) {
    if (i + 1 < nkb) {
      stage((i + 1) & 1, (i + 1) * 32);            // next chunk's DMA in flight
      asm volatile("s_wait_asynccnt 0xa" ::: "memory");   // own part of chunk i landed
    } else {
      asm volatile("s_wait_asynccnt 0x0" ::: "memory");
    }
    __syncthreads();                               // whole chunk landed

    const unsigned short* kt = sK[i & 1];
    const unsigned short* vt = sV[i & 1];

    // ---- S = Q*K^T: B-frags from LDS (2 x ds_load_b128 each) ----
    v8f sfr[2];
    sfr[0] = vzero8(); sfr[1] = vzero8();
    #pragma unroll
    for (int j = 0; j < 2; ++j) {
      const unsigned short* krow = kt + (j * 16 + l16) * Q_HEAD;
      #pragma unroll
      for (int t = 0; t < 6; ++t) {
        v16bf b = cat8(lds8(krow + t * 32 + h * 8), lds8(krow + t * 32 + h * 8 + 16));
        sfr[j] = WMMA_BF16(aq[t], b, sfr[j]);
      }
    }
    #pragma unroll
    for (int j = 0; j < 2; ++j)
      #pragma unroll
      for (int r = 0; r < 8; ++r)
        sS[w][(h * 8 + r) * 32 + j * 16 + l16] = sfr[j][r] * scale;
    asm volatile("s_wait_dscnt 0" ::: "memory");

    // ---- online softmax: lanes 0-15 each own one query row ----
    if (lane < 16) {
      const int row = lane;
      float cmax = -1e30f;
      #pragma unroll
      for (int c = 0; c < 32; ++c) cmax = fmaxf(cmax, sS[w][row * 32 + c]);
      const float m_new = fmaxf(m_run, cmax);
      const float corr  = __expf(m_run - m_new);
      float psum = 0.f;
      #pragma unroll
      for (int c = 0; c < 32; ++c) {
        const float p = __expf(sS[w][row * 32 + c] - m_new);
        psum += p;
        sP[w][row * 32 + c] = f32_to_bf16_rne(p);
      }
      l_run = l_run * corr + psum;
      m_run = m_new;
      sC[w][row] = corr;
    }
    asm volatile("s_wait_dscnt 0" ::: "memory");

    #pragma unroll
    for (int r = 0; r < 8; ++r) {
      const float cr = sC[w][h * 8 + r];
      #pragma unroll
      for (int c = 0; c < 8; ++c) oacc[c][r] *= cr;
    }
    const unsigned short* pp = &sP[w][l16 * 32];
    v16bf pfrag = cat8(lds8(pp + h * 8), lds8(pp + 16 + h * 8));

    // ---- O += P*V: B-frags from LDS ----
    #pragma unroll
    for (int c = 0; c < 8; ++c) {
      const unsigned short* pv = vt + (c * 16 + l16) * 32 + h * 8;
      v16bf b = cat8(lds8(pv), lds8(pv + 16));
      oacc[c] = WMMA_BF16(pfrag, b, oacc[c]);
    }
    __syncthreads();                               // reads done before buf reuse
  }

  if (lane < 16) sC[w][lane] = 1.0f / l_run;
  asm volatile("s_wait_dscnt 0" ::: "memory");
  #pragma unroll
  for (int r = 0; r < 8; ++r) {
    const float inv = sC[w][h * 8 + r];
    float* orow = ohead + ((size_t)hh * S_LEN + q0 + h * 8 + r) * V_HEAD;
    #pragma unroll
    for (int c = 0; c < 8; ++c) orow[c * 16 + l16] = oacc[c][r] * inv;
  }
}

// -------------------- head reduction (deterministic) -----------------------
__global__ void k_reduce_heads(const float* __restrict__ ohead, bf16_t* __restrict__ osum) {
  int i = blockIdx.x * blockDim.x + threadIdx.x;
  if (i >= S_LEN * V_HEAD) return;
  float s = 0.f;
  #pragma unroll
  for (int hh = 0; hh < NHEAD; ++hh) s += ohead[(size_t)hh * S_LEN * V_HEAD + i];
  osum[i] = to_bf(s);
}

// ------------------------------- launcher ----------------------------------
extern "C" void kernel_launch(void* const* d_in, const int* in_sizes, int n_in,
                              void* d_out, int out_size, void* d_ws, size_t ws_size,
                              hipStream_t stream) {
  (void)in_sizes; (void)n_in; (void)out_size; (void)ws_size;

  const float* hidden = (const float*)d_in[0];
  const float* Wq_c   = (const float*)d_in[1];
  const float* bq_c   = (const float*)d_in[2];
  const float* Wq_up  = (const float*)d_in[3];
  const float* bq_up  = (const float*)d_in[4];
  const float* Wkv_c  = (const float*)d_in[5];
  const float* bkv_c  = (const float*)d_in[6];
  const float* Wkv_up = (const float*)d_in[7];
  const float* bkv_up = (const float*)d_in[8];
  const float* Wo     = (const float*)d_in[9];
  const float* bo     = (const float*)d_in[10];

  char* ws = (char*)d_ws;
  size_t off = 0;
  auto carve = [&](size_t bytes) -> char* {
    char* p = ws + off;
    off = (off + bytes + 255) & ~((size_t)255);
    return p;
  };

  bf16_t* hbf    = (bf16_t*)carve((size_t)S_LEN * H_DIM * 2);
  bf16_t* wqc    = (bf16_t*)carve((size_t)Q_LORA * H_DIM * 2);
  bf16_t* wqup   = (bf16_t*)carve((size_t)Q_UP_N * Q_LORA * 2);
  bf16_t* wkvc   = (bf16_t*)carve((size_t)KVC_N * H_DIM * 2);
  bf16_t* wkvup  = (bf16_t*)carve((size_t)KV_UP_N * KV_LORA * 2);
  bf16_t* wo     = (bf16_t*)carve((size_t)H_DIM * V_HEAD * 2);
  bf16_t* qc     = (bf16_t*)carve((size_t)S_LEN * Q_LORA * 2);
  bf16_t* qfull  = (bf16_t*)carve((size_t)S_LEN * Q_UP_N * 2);
  bf16_t* kvc    = (bf16_t*)carve((size_t)S_LEN * KVC_N * 2);
  bf16_t* kvu    = (bf16_t*)carve((size_t)S_LEN * KV_UP_N * 2);
  bf16_t* Qbuf   = (bf16_t*)carve((size_t)NHEAD * S_LEN * Q_HEAD * 2);
  bf16_t* Kbuf   = (bf16_t*)carve((size_t)NHEAD * S_LEN * Q_HEAD * 2);
  bf16_t* Vt     = (bf16_t*)carve((size_t)NHEAD * V_HEAD * S_LEN * 2);
  float*  ohead  = (float*) carve((size_t)NHEAD * S_LEN * V_HEAD * 4);
  bf16_t* osum   = (bf16_t*)carve((size_t)S_LEN * V_HEAD * 2);

  auto conv = [&](const float* src, bf16_t* dst, int n) {
    k_f32_to_bf16<<<(n + 255) / 256, 256, 0, stream>>>(src, dst, n);
  };
  auto gemm_bf = [&](const bf16_t* A, int lda, const bf16_t* W, const float* bias,
                     bf16_t* C, int M, int N, int K) {
    int blocks = (M / 128) * (N / 64);
    k_gemm_wmma<false><<<blocks, GEMM_THREADS, 0, stream>>>(A, lda, W, bias, (void*)C, M, N, K);
  };

  // 1) precision conversion (trivial vs 23.3 TB/s)
  conv(hidden, hbf,   S_LEN * H_DIM);
  conv(Wq_c,   wqc,   Q_LORA * H_DIM);
  conv(Wq_up,  wqup,  Q_UP_N * Q_LORA);
  conv(Wkv_c,  wkvc,  KVC_N * H_DIM);
  conv(Wkv_up, wkvup, KV_UP_N * KV_LORA);
  conv(Wo,     wo,    H_DIM * V_HEAD);

  // 2) projection GEMMs (WMMA bf16, f32 accumulate, LDS-staged W)
  gemm_bf(hbf, H_DIM,  wqc,   bq_c,   qc,    S_LEN, Q_LORA,  H_DIM);
  gemm_bf(qc,  Q_LORA, wqup,  bq_up,  qfull, S_LEN, Q_UP_N,  Q_LORA);
  gemm_bf(hbf, H_DIM,  wkvc,  bkv_c,  kvc,   S_LEN, KVC_N,   H_DIM);
  gemm_bf(kvc, KVC_N,  wkvup, bkv_up, kvu,   S_LEN, KV_UP_N, KV_LORA); // A = kv_nope cols

  // 3) per-head layouts
  {
    int nqk = NHEAD * S_LEN * Q_HEAD;
    k_q_layout<<<(nqk + 255) / 256, 256, 0, stream>>>(qfull, Qbuf);
    k_k_layout<<<(nqk + 255) / 256, 256, 0, stream>>>(kvu, kvc, Kbuf);
    int nv = NHEAD * V_HEAD * S_LEN;
    k_v_layout<<<(nv + 255) / 256, 256, 0, stream>>>(kvu, Vt);
  }

  // 4) flash attention: 32 heads x 32 query-blocks, 4 waves/block
  k_attention<<<NHEAD * (S_LEN / 64), ATT_THREADS, 0, stream>>>(Qbuf, Kbuf, Vt, ohead);

  // 5) head sum + output projection (f32 straight to d_out)
  k_reduce_heads<<<(S_LEN * V_HEAD + 255) / 256, 256, 0, stream>>>(ohead, osum);
  {
    int blocks = (S_LEN / 128) * (H_DIM / 64);
    k_gemm_wmma<true><<<blocks, GEMM_THREADS, 0, stream>>>(
        osum, V_HEAD, wo, bo, d_out, S_LEN, H_DIM, V_HEAD);
  }
}
